// CGC_89455578841730
// MI455X (gfx1250) — compile-verified, matrix operand order checked by reading, not run on hardware
//
#include <hip/hip_runtime.h>
#include <hip/hip_bf16.h>

// ---------------------------------------------------------------------------
// Problem constants (match reference)
// ---------------------------------------------------------------------------
#define D_NUM 4
#define N_ES  2
#define N_SH  4
#define DIN   512
#define H     256
#define BATCH 16384
#define NGATE (N_ES + N_SH)            // 6
#define NGSH  (D_NUM * N_ES + N_SH)    // 12
#define NMAT  12                       // 8 spec + 4 shared GEMMs

#define AROW  520                      // bf16 elems per LDS row (512 + 8 pad)
                                       // 1040B row stride -> 4-bank rotation,
                                       // conflict-free ds_load_b128 A reads

typedef __attribute__((ext_vector_type(2)))  float   v2f;
typedef __attribute__((ext_vector_type(8)))  float   v8f;
typedef __attribute__((ext_vector_type(4)))  __bf16  v4bf;
typedef __attribute__((ext_vector_type(8)))  __bf16  v8bf;
typedef __attribute__((ext_vector_type(16))) __bf16  v16bf;

// ---------------------------------------------------------------------------
// Weight prep: split every W (12 matrices of 512x256, k-major) into bf16
// hi/lo halves stored TRANSPOSED as Wt[m][h][k] (k contiguous) so WMMA B
// fragments are contiguous 16B runs per lane. dst index == flat thread index.
// ---------------------------------------------------------------------------
__global__ __launch_bounds__(256)
void CGC_prep_w(const float* __restrict__ W_spec,
                const float* __restrict__ W_sh,
                __bf16*      __restrict__ Wt_hi,
                __bf16*      __restrict__ Wt_lo) {
    int idx = blockIdx.x * 256 + threadIdx.x;   // (m, h, k), k fastest
    int k = idx & (DIN - 1);
    int t = idx >> 9;
    int h = t & (H - 1);
    int m = t >> 8;

    float w = (m < 8) ? W_spec[((size_t)m * DIN + k) * H + h]
                      : W_sh[((size_t)(m - 8) * DIN + k) * H + h];
    __bf16 hi = (__bf16)w;
    __bf16 lo = (__bf16)(w - (float)hi);
    Wt_hi[idx] = hi;
    Wt_lo[idx] = lo;
}

// ---------------------------------------------------------------------------
// Gate kernel 1: g[n,b,:] = softmax(softmax(xd[n,b] @ W_gate[n] + b_gate[n]) + mask[n])
// n comes from blockIdx.y so every W_gate access is wave-uniform (scalar loads).
// ---------------------------------------------------------------------------
__global__ __launch_bounds__(256)
void CGC_gate_spec(const float* __restrict__ x_list,
                   const int*   __restrict__ sim_domain,
                   const float* __restrict__ W_gate,
                   const float* __restrict__ b_gate,
                   float*       __restrict__ g_out) {
    const int n = blockIdx.y;
    const int b = blockIdx.x * blockDim.x + threadIdx.x;
    if (b >= BATCH) return;

    const float* x  = x_list + ((size_t)n * BATCH + b) * DIN;
    const float* Wg = W_gate + (size_t)n * DIN * NGATE;

    float acc[NGATE];
#pragma unroll
    for (int e = 0; e < NGATE; ++e) acc[e] = b_gate[n * NGATE + e];

    for (int k = 0; k < DIN; k += 4) {
        float4 xv = *(const float4*)(x + k);
        const float* w = Wg + (size_t)k * NGATE;
#pragma unroll
        for (int e = 0; e < NGATE; ++e) {
            acc[e] += xv.x * w[e] + xv.y * w[NGATE + e]
                    + xv.z * w[2 * NGATE + e] + xv.w * w[3 * NGATE + e];
        }
    }

    float m = acc[0];
#pragma unroll
    for (int e = 1; e < NGATE; ++e) m = fmaxf(m, acc[e]);
    float s = 0.f;
    float p[NGATE];
#pragma unroll
    for (int e = 0; e < NGATE; ++e) { p[e] = expf(acc[e] - m); s += p[e]; }
    float inv = 1.f / s;
#pragma unroll
    for (int e = 0; e < NGATE; ++e) p[e] *= inv;

    int s0 = sim_domain[n * 2 + 0];
    int s1 = sim_domain[n * 2 + 1];
#pragma unroll
    for (int d = 0; d < N_SH; ++d) {
        bool member = (s0 == d) || (s1 == d);
        if (!member) p[N_ES + d] = -1.0e30f;  // exp() underflows to exactly 0
    }

    m = p[0];
#pragma unroll
    for (int e = 1; e < NGATE; ++e) m = fmaxf(m, p[e]);
    s = 0.f;
#pragma unroll
    for (int e = 0; e < NGATE; ++e) { p[e] = expf(p[e] - m); s += p[e]; }
    inv = 1.f / s;

    float* go = g_out + ((size_t)n * BATCH + b) * NGATE;
#pragma unroll
    for (int e = 0; e < NGATE; ++e) go[e] = p[e] * inv;
}

// ---------------------------------------------------------------------------
// Gate kernel 2: gs[b,:] = softmax(xs[b] @ W_gate_sh + b_gate_sh)
// ---------------------------------------------------------------------------
__global__ __launch_bounds__(256)
void CGC_gate_sh(const float* __restrict__ x_list,
                 const float* __restrict__ W_gate_sh,
                 const float* __restrict__ b_gate_sh,
                 float*       __restrict__ gs_out) {
    const int b = blockIdx.x * blockDim.x + threadIdx.x;
    if (b >= BATCH) return;

    const float* x = x_list + ((size_t)D_NUM * BATCH + b) * DIN;  // xs = x_list[-1]

    float acc[NGSH];
#pragma unroll
    for (int e = 0; e < NGSH; ++e) acc[e] = b_gate_sh[e];

    for (int k = 0; k < DIN; k += 4) {
        float4 xv = *(const float4*)(x + k);
        const float* w = W_gate_sh + (size_t)k * NGSH;
#pragma unroll
        for (int e = 0; e < NGSH; ++e) {
            acc[e] += xv.x * w[e] + xv.y * w[NGSH + e]
                    + xv.z * w[2 * NGSH + e] + xv.w * w[3 * NGSH + e];
        }
    }

    float m = acc[0];
#pragma unroll
    for (int e = 1; e < NGSH; ++e) m = fmaxf(m, acc[e]);
    float s = 0.f;
#pragma unroll
    for (int e = 0; e < NGSH; ++e) { acc[e] = expf(acc[e] - m); s += acc[e]; }
    float inv = 1.f / s;

    float* go = gs_out + (size_t)b * NGSH;
#pragma unroll
    for (int e = 0; e < NGSH; ++e) go[e] = acc[e] * inv;
}

// ---------------------------------------------------------------------------
// Fused main kernel (bf16x3 split-precision WMMA).
//
// Block = 256 threads (8 waves) owns a 16-row batch slab; wave w owns H
// columns [32w, 32w+32) (two 16x16 N-tiles). GEMMs grouped by x-plane:
//   grp 0..3: xd[grp] -> spec GEMMs m = 2*grp, 2*grp+1
//   grp 4   : xs      -> shared GEMMs m = 8..11
// Per group the 16x512 x-tile is split to bf16 hi/lo during LDS staging.
// Per K=32 step and N-tile, three V_WMMA_F32_16X16X32_BF16 accumulate
// hi*HI + hi*LO + lo*HI into the f32 C tile (~2^-17 input accuracy).
// Each relu'd GEMM tile folds straight into 5 gate-weighted accumulators.
//
// Fragment layouts (wave32, ISA 7.12.2, 16-bit A 16x32 / B 32x16):
//   per lane: elems 0..7  <-> K = k + 8*hiHalf + (0..7)
//             elems 8..15 <-> K = k + 16 + 8*hiHalf + (0..7)
//   A: M = lane%16 (row of x-tile);  B: N = lane%16 (column of W)
//   C (16x16, 8 VGPR): row = vgpr_j + 8*hiHalf, col = lane%16
// ---------------------------------------------------------------------------
__global__ __launch_bounds__(256)
void CGC_moe_main(const float*  __restrict__ x_list,
                  const __bf16* __restrict__ Wt_hi,
                  const __bf16* __restrict__ Wt_lo,
                  const float*  __restrict__ b_spec,
                  const float*  __restrict__ b_sh,
                  const float*  __restrict__ g,
                  const float*  __restrict__ gs,
                  float*        __restrict__ out) {
    __shared__ __align__(16) __bf16 ldsA[2][16 * AROW];   // hi, lo: 33 KB total

    const int tid     = threadIdx.x;
    const int lane    = tid & 31;
    const int wave    = tid >> 5;           // 0..7 -> covers full H
    const int r0      = blockIdx.x * 16;    // batch-row base
    const int c0      = wave * 32;          // H-column base (two 16-wide tiles)
    const int lane16  = lane & 15;
    const int hiHalf  = lane >> 4;

    // loop-invariant per-lane element offsets
    const int aoff = lane16 * AROW + hiHalf * 8;          // A fragment in LDS
    const int boff = (c0 + lane16) * DIN + hiHalf * 8;    // B fragment in Wt[m]
    const int grow = r0 + 8 * hiHalf;                     // gate row base

    v8f acc[5][2];
#pragma unroll
    for (int o = 0; o < 5; ++o) {
        acc[o][0] = (v8f){};
        acc[o][1] = (v8f){};
    }

    for (int grp = 0; grp < 5; ++grp) {
        // ---- stage + bf16-split this group's 16x512 x-tile into LDS ----
        __syncthreads();                    // previous group done reading
#pragma unroll
        for (int c = 0; c < 8; ++c) {       // 16*128 float4 chunks / 256 threads
            int chunk = tid + c * 256;
            int row   = chunk >> 7;
            int col4  = (chunk & 127) << 2;
            float4 v = *(const float4*)(x_list +
                        ((size_t)grp * BATCH + (r0 + row)) * DIN + col4);
            __bf16 h0 = (__bf16)v.x, h1 = (__bf16)v.y,
                   h2 = (__bf16)v.z, h3 = (__bf16)v.w;
            v4bf hi = {h0, h1, h2, h3};
            v4bf lo = {(__bf16)(v.x - (float)h0), (__bf16)(v.y - (float)h1),
                       (__bf16)(v.z - (float)h2), (__bf16)(v.w - (float)h3)};
            *(v4bf*)&ldsA[0][row * AROW + col4] = hi;
            *(v4bf*)&ldsA[1][row * AROW + col4] = lo;
        }
        __syncthreads();                    // tile visible to all waves

        const int nGem = (grp == 4) ? 4 : 2;
        for (int q = 0; q < nGem; ++q) {
            const int m = (grp == 4) ? (8 + q) : (grp * 2 + q);
            const __bf16* Wh = Wt_hi + (size_t)m * (H * DIN);
            const __bf16* Wl = Wt_lo + (size_t)m * (H * DIN);
            const float* bias = (grp == 4) ? (b_sh + q * H) : (b_spec + m * H);

            v8f ct0 = (v8f){};
            v8f ct1 = (v8f){};

#pragma unroll 2
            for (int k = 0; k < DIN; k += 32) {
                // A fragments (hi/lo) from LDS: 4x ds_load_b128
                v8bf ah0 = *(const v8bf*)&ldsA[0][aoff + k];
                v8bf ah1 = *(const v8bf*)&ldsA[0][aoff + k + 16];
                v8bf al0 = *(const v8bf*)&ldsA[1][aoff + k];
                v8bf al1 = *(const v8bf*)&ldsA[1][aoff + k + 16];
                v16bf a_hi = __builtin_shufflevector(ah0, ah1,
                    0,1,2,3,4,5,6,7,8,9,10,11,12,13,14,15);
                v16bf a_lo = __builtin_shufflevector(al0, al1,
                    0,1,2,3,4,5,6,7,8,9,10,11,12,13,14,15);

                // B fragments (hi/lo, two N-tiles): 8x global_load_b128
                int wo = boff + k;
                v8bf bh00 = *(const v8bf*)(Wh + wo);
                v8bf bh01 = *(const v8bf*)(Wh + wo + 16);
                v8bf bh10 = *(const v8bf*)(Wh + wo + 16 * DIN);
                v8bf bh11 = *(const v8bf*)(Wh + wo + 16 * DIN + 16);
                v8bf bl00 = *(const v8bf*)(Wl + wo);
                v8bf bl01 = *(const v8bf*)(Wl + wo + 16);
                v8bf bl10 = *(const v8bf*)(Wl + wo + 16 * DIN);
                v8bf bl11 = *(const v8bf*)(Wl + wo + 16 * DIN + 16);
                v16bf b0h = __builtin_shufflevector(bh00, bh01,
                    0,1,2,3,4,5,6,7,8,9,10,11,12,13,14,15);
                v16bf b1h = __builtin_shufflevector(bh10, bh11,
                    0,1,2,3,4,5,6,7,8,9,10,11,12,13,14,15);
                v16bf b0l = __builtin_shufflevector(bl00, bl01,
                    0,1,2,3,4,5,6,7,8,9,10,11,12,13,14,15);
                v16bf b1l = __builtin_shufflevector(bl10, bl11,
                    0,1,2,3,4,5,6,7,8,9,10,11,12,13,14,15);

                // bf16x3: hi*HI + hi*LO + lo*HI (f32 accumulate)
                ct0 = __builtin_amdgcn_wmma_f32_16x16x32_bf16(
                          false, a_hi, false, b0h, (short)0, ct0, false, false);
                ct0 = __builtin_amdgcn_wmma_f32_16x16x32_bf16(
                          false, a_hi, false, b0l, (short)0, ct0, false, false);
                ct0 = __builtin_amdgcn_wmma_f32_16x16x32_bf16(
                          false, a_lo, false, b0h, (short)0, ct0, false, false);
                ct1 = __builtin_amdgcn_wmma_f32_16x16x32_bf16(
                          false, a_hi, false, b1h, (short)0, ct1, false, false);
                ct1 = __builtin_amdgcn_wmma_f32_16x16x32_bf16(
                          false, a_hi, false, b1l, (short)0, ct1, false, false);
                ct1 = __builtin_amdgcn_wmma_f32_16x16x32_bf16(
                          false, a_lo, false, b1h, (short)0, ct1, false, false);
            }

            // bias + relu (register epilogue)
            float bias0 = bias[c0 + lane16];
            float bias1 = bias[c0 + 16 + lane16];
#pragma unroll
            for (int j = 0; j < 8; ++j) {
                float v0 = ct0[j] + bias0;
                float v1 = ct1[j] + bias1;
                ct0[j] = v0 > 0.f ? v0 : 0.f;
                ct1[j] = v1 > 0.f ? v1 : 0.f;
            }

            // gate-weighted accumulation into the 5 output planes
            if (m < 8) {
                const int n = m >> 1, e = m & 1;
                const float* gn = g + (size_t)n * BATCH * NGATE;
#pragma unroll
                for (int j = 0; j < 8; ++j) {
                    float wg  = gn[(grow + j) * NGATE + e];
                    float wgs = gs[(grow + j) * NGSH + m];
                    acc[n][0][j] += wg  * ct0[j];
                    acc[n][1][j] += wg  * ct1[j];
                    acc[4][0][j] += wgs * ct0[j];
                    acc[4][1][j] += wgs * ct1[j];
                }
            } else {
                const int e = m - 8;
#pragma unroll
                for (int j = 0; j < 8; ++j) {
                    float wgs = gs[(grow + j) * NGSH + D_NUM * N_ES + e];
                    acc[4][0][j] += wgs * ct0[j];
                    acc[4][1][j] += wgs * ct1[j];
#pragma unroll
                    for (int n = 0; n < D_NUM; ++n) {
                        float wg = g[((size_t)n * BATCH + (grow + j)) * NGATE
                                     + N_ES + e];
                        acc[n][0][j] += wg * ct0[j];
                        acc[n][1][j] += wg * ct1[j];
                    }
                }
            }
        }
    }

    // store: out is (5, BATCH, H) fp32; 32-bit offsets keep saddr form
#pragma unroll
    for (int o = 0; o < 5; ++o) {
#pragma unroll
        for (int t = 0; t < 2; ++t) {
            const int colbase = c0 + t * 16 + lane16;
#pragma unroll
            for (int j = 0; j < 8; ++j) {
                int off = (o * BATCH + (grow + j)) * H + colbase;
                out[off] = acc[o][t][j];
            }
        }
    }
}

// ---------------------------------------------------------------------------
// Launch
// ---------------------------------------------------------------------------
extern "C" void kernel_launch(void* const* d_in, const int* in_sizes, int n_in,
                              void* d_out, int out_size, void* d_ws, size_t ws_size,
                              hipStream_t stream) {
    const float* x_list     = (const float*)d_in[0];
    const int*   sim_domain = (const int*)  d_in[1];
    const float* W_spec     = (const float*)d_in[2];
    const float* b_spec     = (const float*)d_in[3];
    const float* W_sh       = (const float*)d_in[4];
    const float* b_sh       = (const float*)d_in[5];
    const float* W_gate     = (const float*)d_in[6];
    const float* b_gate     = (const float*)d_in[7];
    const float* W_gate_sh  = (const float*)d_in[8];
    const float* b_gate_sh  = (const float*)d_in[9];
    float* out = (float*)d_out;

    // workspace layout: Wt_hi | Wt_lo (bf16, transposed splits) | g | gs (f32)
    const size_t wElems = (size_t)NMAT * H * DIN;         // 1,572,864
    __bf16* Wt_hi = (__bf16*)d_ws;
    __bf16* Wt_lo = Wt_hi + wElems;
    float*  g     = (float*)(Wt_lo + wElems);
    float*  gs    = g + (size_t)D_NUM * BATCH * NGATE;

    CGC_prep_w<<<(int)(wElems / 256), 256, 0, stream>>>(
        W_spec, W_sh, Wt_hi, Wt_lo);

    dim3 gateGrid(BATCH / 256, D_NUM);
    CGC_gate_spec<<<gateGrid, 256, 0, stream>>>(
        x_list, sim_domain, W_gate, b_gate, g);
    CGC_gate_sh<<<BATCH / 256, 256, 0, stream>>>(
        x_list, W_gate_sh, b_gate_sh, gs);

    CGC_moe_main<<<BATCH / 16, 256, 0, stream>>>(
        x_list, Wt_hi, Wt_lo, b_spec, b_sh, g, gs, out);
}
